// EDGE_NET_90013924590246
// MI455X (gfx1250) — matrix operand order repaired and checked by reading, not compiled, not run on hardware
//
#include <hip/hip_runtime.h>
#include <hip/hip_bf16.h>

// MI455X / gfx1250 implementation.
// Pipeline:
//   1) 3-pass prefix scan over boundary flags of sorted src_ids -> gid[N],
//      group_start[], last_gid (all in d_ws).
//   2) Per-group column means (atomic-free: groups are contiguous ranges),
//      written into the imputed_out region of d_out (acts as imp_second).
//   3) Two bf16-WMMA tiled GEMMs (v_wmma_f32_16x16x32_bf16):
//        x_out       = [x_src | embs[gid]] @ W1^T + b1   (N x 128)
//        imputed_out = [embs  | imp2     ] @ W2^T + b2   (G x 128)
//      GEMM2 overwrites imp2 in place (safe: A fully staged per block
//      before any output store; blocks own disjoint row ranges).
//
// GEMM staging: vectorized (global_load_b128 -> v_cvt_pk_bf16_f32 ->
// ds_store_b64 into pre-swizzled WMMA fragment layout) and software-
// pipelined: chunk kt+1's loads are issued before chunk kt's WMMAs so the
// results stay live across the barrier, forcing distinct destination
// registers (batched issue, one wait hidden behind 8 WMMAs). All address
// math uses 32-bit offsets so the backend can use saddr+voffset addressing.

typedef __attribute__((ext_vector_type(16))) __bf16 v16bf;
typedef __attribute__((ext_vector_type(4)))  __bf16 v4bf;
typedef __attribute__((ext_vector_type(8)))  float  v8f;

#define HDIM 128           // H
#define KDIM 256           // 2H

// ---------------------------------------------------------------- scan pass 1
__global__ void EDGE_scan1(const int* __restrict__ ids, int n, int* __restrict__ bsum) {
    int t = threadIdx.x;
    int base = blockIdx.x * 1024 + t * 4;
    int s = 0;
    #pragma unroll
    for (int j = 0; j < 4; ++j) {
        int i = base + j;
        if (i < n && i > 0 && ids[i] != ids[i - 1]) s++;
    }
    __shared__ int red[256];
    red[t] = s;
    __syncthreads();
    for (int off = 128; off > 0; off >>= 1) {
        if (t < off) red[t] += red[t + off];
        __syncthreads();
    }
    if (t == 0) bsum[blockIdx.x] = red[0];
}

// ---------------------------------------------------------------- scan pass 2
__global__ void EDGE_scan2(const int* __restrict__ bsum, int nb, int* __restrict__ boff) {
    __shared__ int lds[1024];
    int t = threadIdx.x;
    int v = (t < nb) ? bsum[t] : 0;
    lds[t] = v;
    __syncthreads();
    for (int off = 1; off < 1024; off <<= 1) {
        int x = (t >= off) ? lds[t - off] : 0;
        __syncthreads();
        lds[t] += x;
        __syncthreads();
    }
    if (t < nb) boff[t] = lds[t] - v;   // exclusive block offsets
}

// ---------------------------------------------------------------- scan pass 3
__global__ void EDGE_scan3(const int* __restrict__ ids, int n,
                           const int* __restrict__ boff,
                           int* __restrict__ gid,
                           int* __restrict__ group_start,
                           int* __restrict__ lastgid) {
    int t = threadIdx.x;
    int base = blockIdx.x * 1024 + t * 4;
    int f[4];
    int s = 0;
    #pragma unroll
    for (int j = 0; j < 4; ++j) {
        int i = base + j;
        f[j] = (i < n && i > 0 && ids[i] != ids[i - 1]) ? 1 : 0;
        s += f[j];
    }
    __shared__ int lds[256];
    lds[t] = s;
    __syncthreads();
    for (int off = 1; off < 256; off <<= 1) {
        int x = (t >= off) ? lds[t - off] : 0;
        __syncthreads();
        lds[t] += x;
        __syncthreads();
    }
    int run = lds[t] - s + boff[blockIdx.x];   // exclusive prefix for this thread
    #pragma unroll
    for (int j = 0; j < 4; ++j) {
        int i = base + j;
        if (i < n) {
            run += f[j];
            gid[i] = run;
            if (f[j] || i == 0) group_start[run] = i;
            if (i == n - 1) { lastgid[0] = run; group_start[run + 1] = n; }
        }
    }
}

// ------------------------------------------------------- per-group column mean
__global__ void EDGE_group_mean(const float* __restrict__ x_src,
                                const float* __restrict__ embs,
                                const int* __restrict__ group_start,
                                const int* __restrict__ lastgid_p,
                                float* __restrict__ imp2) {
    int g = blockIdx.x;
    int h = threadIdx.x;                // 128 threads, one column each
    int lastg = lastgid_p[0];
    int go = g * HDIM + h;              // < 6.4M, int is fine
    if (g >= lastg) {                   // 'written' is strictly g < gid[-1]
        imp2[go] = embs[go];
        return;
    }
    int s = group_start[g];
    int e = group_start[g + 1];
    float sum = 0.f;
    for (int r = s; r < e; ++r) sum += x_src[r * HDIM + h];   // < 64M, int ok
    float cnt = (float)(e - s);
    imp2[go] = sum / fmaxf(cnt, 1.0f);
}

// -------------------------------------------------------------- WMMA GEMM
// out[i][o] = sum_k Acat[i][k] * W[o][k] + bias[o]
//   Acat[i][0:128)   = A_left[i]
//   Acat[i][128:256) = A_right[ridx ? ridx[i] : i]
// W is [128][256] row-major (reference W.T GEMM). M-tile = 128 rows/block.
__global__ __launch_bounds__(256)
void EDGE_gemm_wmma(const float* __restrict__ A_left,
                    const float* __restrict__ A_right,
                    const int*   __restrict__ ridx,
                    const float* __restrict__ W,
                    const float* __restrict__ bias,
                    float* __restrict__ out,
                    int M) {
    // Fragment-layout LDS buffers: [tile][lane][16 bf16] -> each lane's
    // fragment is 32 contiguous bytes (2x ds_load_b128).
    __shared__ __align__(32) __bf16 Abuf[8][32][16];   // 8 wave-row-groups x 32 K
    __shared__ __align__(32) __bf16 Bbuf[8][32][16];   // 8 col-tiles x 32 K
    __bf16* const Aflat = &Abuf[0][0][0];
    __bf16* const Bflat = &Bbuf[0][0][0];

    const int t    = threadIdx.x;
    const int lane = t & 31;
    const int w    = t >> 5;            // wave id == 16-row stripe id
    const int rowBase = blockIdx.x * 128;

    // ---- precompute kt-invariant staging state (4 vec4 slots per thread)
    // slot s covers (r = vec>>3, kc = (vec&7)*4) with vec = t + 256*s; the
    // same decomposition serves A (row r) and B (column n = r).
    int  aEL[4];                        // 32-bit elem offset into A_left
    int  aER[4];                        // 32-bit elem offset into A_right
    int  bEW[4];                        // 32-bit elem offset into W
    bool aOk[4];                        // row in range
    int  aOff[4];                       // LDS element offset for A slot
    int  bOff[4];                       // LDS element offset for B slot
    #pragma unroll
    for (int s = 0; s < 4; ++s) {
        int vec = t + 256 * s;
        int r   = vec >> 3;
        int kc  = (vec & 7) * 4;
        int gr  = rowBase + r;
        bool ok = gr < M;
        int grc = ok ? gr : (M - 1);
        aOk[s] = ok;
        int ri  = ridx ? ridx[grc] : grc;
        aEL[s] = grc * HDIM + kc;       // max 64M elems -> int ok
        aER[s] = ri  * HDIM + kc;
        bEW[s] = r   * KDIM + kc;       // n == r
        // A fragment map (16-bit A 16x32, ISA 7.12.2): 4 consecutive K -> 4
        // contiguous jj in one lane.
        int m = r & 15, wr = r >> 4;
        int sub = kc & 15, half = kc >> 4;
        int l2 = m + ((sub & 8) ? 16 : 0);
        int jj = (sub & 7) + 8 * half;
        aOff[s] = (wr * 32 + l2) * 16 + jj;
        // B fragment map (16-bit B 32x16): same K order as A (lane half by K bit 3).
        int nt = r >> 4;
        int bl2 = (r & 15) + ((kc & 8) ? 16 : 0);
        bOff[s] = (nt * 32 + bl2) * 16 + (kc & 7) + 8 * (kc >> 4);
    }

    // issue all 8 global b128 loads for chunk kt (no waits between them)
    auto issueLoads = [&](int kt, float4* va, float4* vb) {
        const bool left = kt < 4;       // chunk never straddles the concat
        const float* __restrict__ Asrc = left ? A_left : A_right;
        const int kb = (kt & 3) * 32;   // column base within the source half
        #pragma unroll
        for (int s = 0; s < 4; ++s)
            va[s] = *(const float4*)(Asrc + (left ? aEL[s] : aER[s]) + kb);
        #pragma unroll
        for (int s = 0; s < 4; ++s)
            vb[s] = *(const float4*)(W + bEW[s] + kt * 32);
    };

    v8f acc[8] = {};                    // 8 col-tiles of 16x16 f32

    float4 va[4], vb[4];
    issueLoads(0, va, vb);

    for (int kt = 0; kt < 8; ++kt) {    // K = 256 in chunks of 32
        __syncthreads();                // prior-iteration frag reads done

        // ---- convert + pack current chunk into fragment LDS (ds_store_b64)
        #pragma unroll
        for (int s = 0; s < 4; ++s) {
            float4 vA = va[s];
            if (!aOk[s]) { vA.x = 0.f; vA.y = 0.f; vA.z = 0.f; vA.w = 0.f; }
            v4bf ca;
            ca[0] = (__bf16)vA.x; ca[1] = (__bf16)vA.y;
            ca[2] = (__bf16)vA.z; ca[3] = (__bf16)vA.w;
            *(v4bf*)&Aflat[aOff[s]] = ca;

            float4 vB = vb[s];
            v4bf cb;
            cb[0] = (__bf16)vB.x; cb[1] = (__bf16)vB.y;
            cb[2] = (__bf16)vB.z; cb[3] = (__bf16)vB.w;
            *(v4bf*)&Bflat[bOff[s]] = cb;
        }

        // ---- issue next chunk's loads now; results stay live across the
        // barrier + WMMA phase -> distinct registers, batched issue.
        float4 na[4], nb[4];
        if (kt < 7) {
            issueLoads(kt + 1, na, nb);
            // prefetch two chunks ahead (global_prefetch_b8)
            if (kt < 6) {
                const bool pl = (kt + 2) < 4;
                const float* __restrict__ pA = pl ? A_left : A_right;
                const int pkb = ((kt + 2) & 3) * 32;
                #pragma unroll
                for (int s = 0; s < 4; ++s)
                    __builtin_prefetch(pA + (pl ? aEL[s] : aER[s]) + pkb, 0, 1);
            }
        }

        __syncthreads();

        // ---- compute: 8 WMMAs per wave per chunk
        v16bf afrag = *(const v16bf*)&Abuf[w][lane][0];
        #pragma unroll
        for (int nt = 0; nt < 8; ++nt) {
            v16bf bfrag = *(const v16bf*)&Bbuf[nt][lane][0];
            acc[nt] = __builtin_amdgcn_wmma_f32_16x16x32_bf16(
                /*neg_a=*/false, afrag, /*neg_b=*/false, bfrag,
                /*c_mod=*/(short)0, acc[nt],
                /*reuse_a=*/false, /*reuse_b=*/false);
        }

        if (kt < 7) {
            #pragma unroll
            for (int s = 0; s < 4; ++s) { va[s] = na[s]; vb[s] = nb[s]; }
        }
    }

    // ---- epilogue: C/D layout (VGPR v -> row v + 8*laneHi, col = laneLo)
    const int laneLo = lane & 15;
    const int laneHi = lane >> 4;
    #pragma unroll
    for (int nt = 0; nt < 8; ++nt) {
        int c = nt * 16 + laneLo;
        float bv = bias[c];
        #pragma unroll
        for (int v = 0; v < 8; ++v) {
            int gr = rowBase + w * 16 + v + 8 * laneHi;
            if (gr < M) out[gr * HDIM + c] = acc[nt][v] + bv;   // < 64M, int ok
        }
    }
}

// ----------------------------------------------------------------- launcher
extern "C" void kernel_launch(void* const* d_in, const int* in_sizes, int n_in,
                              void* d_out, int out_size, void* d_ws, size_t ws_size,
                              hipStream_t stream) {
    const float* x_src = (const float*)d_in[0];   // [N,128]
    const float* embs  = (const float*)d_in[1];   // [G,128]
    const float* W1    = (const float*)d_in[2];   // [128,256]
    const float* b1    = (const float*)d_in[3];   // [128]
    const float* W2    = (const float*)d_in[4];   // [128,256]
    const float* b2    = (const float*)d_in[5];   // [128]
    const int*   ids   = (const int*)d_in[6];     // [N] sorted

    const int N = in_sizes[6];
    const int G = in_sizes[1] / HDIM;

    // workspace layout (ints)
    int* gid         = (int*)d_ws;                 // N
    int* group_start = gid + N;                    // G+1
    int* bsum        = group_start + (G + 1);      // <=1024
    int* boff        = bsum + 1024;                // <=1024
    int* lastgid     = boff + 1024;                // 1

    float* x_out = (float*)d_out;                  // [N,128]
    float* imp2  = x_out + (size_t)N * HDIM;       // [G,128]; mean then output

    const int nb = (N + 1023) / 1024;              // scan blocks (489 for N=500k)

    EDGE_scan1<<<nb, 256, 0, stream>>>(ids, N, bsum);
    EDGE_scan2<<<1, 1024, 0, stream>>>(bsum, nb, boff);
    EDGE_scan3<<<nb, 256, 0, stream>>>(ids, N, boff, gid, group_start, lastgid);

    EDGE_group_mean<<<G, HDIM, 0, stream>>>(x_src, embs, group_start, lastgid, imp2);

    // x_out = [x_src | embs[gid]] @ W1^T + b1
    EDGE_gemm_wmma<<<(N + 127) / 128, 256, 0, stream>>>(
        x_src, embs, gid, W1, b1, x_out, N);

    // imputed_out = [embs | imp2] @ W2^T + b2   (overwrites imp2 region safely)
    EDGE_gemm_wmma<<<(G + 127) / 128, 256, 0, stream>>>(
        embs, imp2, nullptr, W2, b2, imp2, G);
}
